// Memory_trans_read_38079180046960
// MI455X (gfx1250) — matrix operand order repaired and checked by reading, not compiled, not run on hardware
//
#include <hip/hip_runtime.h>

// CDNA5 / gfx1250, wave32. Flash-attention-style memory read:
//   score = Q·Kᵀ (bf16 WMMA, f32 acc) -> online softmax over m -> out = P·K
// Computed transposed (Sᵀ, Outᵀ) so softmax + P re-layout are per-lane
// (single xor-16 shuffle). Double-buffered LDS staging: next chunk's global
// loads are issued before the WMMA loops (LOADcnt) and converted/stored to
// LDS after them, keeping DScnt clean for the pipelined WMMA operand loads.

typedef __attribute__((ext_vector_type(16))) __bf16 v16bf;
typedef __attribute__((ext_vector_type(8)))  __bf16 v8bf;
typedef __attribute__((ext_vector_type(4)))  __bf16 v4bf;
typedef __attribute__((ext_vector_type(2)))  __bf16 v2bf;
typedef __attribute__((ext_vector_type(8)))  float  v8f;

#define D_DIM         256
#define M_KEYS        2048
#define CHUNK         32      // keys per main-loop iteration (two 16-key WMMA tiles)
#define HW            4096    // H*W
#define ROWS_PER_WAVE 16
#define WAVES_PER_WG  4
#define ROWS_PER_WG   (ROWS_PER_WAVE * WAVES_PER_WG)   // 64
#define LOG2E         1.4426950408889634f

static __device__ __forceinline__ __bf16 f2bf_sw(float f) {
  union { float f; unsigned u; } in; in.f = f;
  const unsigned u = in.u;
  const unsigned r = u + 0x7FFFu + ((u >> 16) & 1u);   // round-to-nearest-even
  union { unsigned short s; __bf16 b; } o; o.s = (unsigned short)(r >> 16);
  return o.b;
}

#if __has_builtin(__builtin_amdgcn_cvt_pk_bf16_f32)
static __device__ __forceinline__ v2bf f2bf2(float a, float b) {
  auto t = __builtin_amdgcn_cvt_pk_bf16_f32(a, b);     // v_cvt_pk_bf16_f32
  return __builtin_bit_cast(v2bf, t);
}
#else
static __device__ __forceinline__ v2bf f2bf2(float a, float b) {
  v2bf r; r[0] = f2bf_sw(a); r[1] = f2bf_sw(b); return r;
}
#endif

static __device__ __forceinline__ v8f wmma_bf16(v16bf a, v16bf b, v8f c) {
  return __builtin_amdgcn_wmma_f32_16x16x32_bf16(false, a, false, b,
                                                 (short)0, c, false, false);
}

__global__ __launch_bounds__(128, 1)
void mem_attn_read_kernel(const float* __restrict__ keys,
                          const float* __restrict__ query,
                          float* __restrict__ out)
{
  // Double-buffered key chunk in LDS as bf16, in BOTH layouts:
  //  kRowB[p][key][d] -> A operand of the score WMMA (keys row-major)
  //  kTB  [p][d][key] -> A operand of the output WMMA (keys transposed)
  __shared__ __bf16 kRowB[2][CHUNK * D_DIM];   // 2 x 16 KB
  __shared__ __bf16 kTB[2][D_DIM * CHUNK];     // 2 x 16 KB

  const int tid  = threadIdx.x;
  const int lane = tid & 31;
  const int wave = tid >> 5;
  const int h    = lane >> 4;     // lane half (K-split in WMMA layouts)
  const int ln   = lane & 15;
  const int kk   = tid >> 2;      // staging: key within chunk (0..31)
  const int d0s  = (tid & 3) * 64;

  const int wave_row0 = blockIdx.x * ROWS_PER_WG + wave * ROWS_PER_WAVE;

  // ---------------- resident Qᵀ fragments (B operand of Sᵀ WMMA) -------------
  // B 32x16 bf16 layout: lane(h,n) elems 0..7 -> K=8h+e, elems 8..15 -> K=16+8h+(e-8).
  // query is NCHW: element (row, d) at qb*(D*HW) + d*HW + hw. One-time gather.
  const int qr  = wave_row0 + ln;
  const int qb  = qr >> 12;             // / 4096
  const int qhw = qr & (HW - 1);
  const float* qcol = query + (size_t)qb * (D_DIM * HW) + qhw;

  v16bf qfrag[8];
#pragma unroll
  for (int c = 0; c < 8; ++c) {
#pragma unroll
    for (int i = 0; i < 8; ++i) {
      const int d = c * 32 + h * 8 + 2 * i + (i >= 4 ? 8 : 0);
      const v2bf pr = f2bf2(qcol[(size_t)d * HW], qcol[(size_t)(d + 1) * HW]);
      qfrag[c][2 * i]     = pr[0];
      qfrag[c][2 * i + 1] = pr[1];
    }
  }

  v8f zero;
#pragma unroll
  for (int i = 0; i < 8; ++i) zero[i] = 0.0f;

  v8f acc[16];                     // Outᵀ accumulators: 16 d-tiles of 16x16 f32
#pragma unroll
  for (int t = 0; t < 16; ++t) acc[t] = zero;

  float run_max = -1e30f;
  float run_sum = 0.0f;

  // ---------------- stage chunk 0 into buffer 0 ------------------------------
  {
    const float* src = keys + (size_t)kk * D_DIM + d0s;
#pragma unroll
    for (int i = 0; i < 16; ++i) {
      const float4 v = ((const float4*)src)[i];
      const int d = d0s + 4 * i;
      const v2bf c01 = f2bf2(v.x, v.y);
      const v2bf c23 = f2bf2(v.z, v.w);
      v4bf row; row[0] = c01[0]; row[1] = c01[1]; row[2] = c23[0]; row[3] = c23[1];
      *(v4bf*)&kRowB[0][kk * D_DIM + d] = row;
      kTB[0][(d + 0) * CHUNK + kk] = c01[0];
      kTB[0][(d + 1) * CHUNK + kk] = c01[1];
      kTB[0][(d + 2) * CHUNK + kk] = c23[0];
      kTB[0][(d + 3) * CHUNK + kk] = c23[1];
    }
  }

  int p = 0;
  for (int kb = 0; kb < M_KEYS; kb += CHUNK) {
    __syncthreads();   // staged data visible; previous buffer free to overwrite

    const __bf16* kRow = kRowB[p];
    const __bf16* kT   = kTB[p];
    const bool more = (kb + CHUNK) < M_KEYS;

    // -------- issue next chunk's global loads now (LOADcnt only) ------------
    float4 g[16];
    if (more) {
      const float* src = keys + (size_t)(kb + CHUNK + kk) * D_DIM + d0s;
#pragma unroll
      for (int i = 0; i < 16; ++i) g[i] = ((const float4*)src)[i];
    }

    // -------- Sᵀ = K_tile(16x32d) x Qᵀ(32d x 16q), pipelined over d ---------
    // A 16x32 bf16 layout: lane(h,row) elems 0..7 -> K=8h.., elems 8..15 -> 16+8h..
    v8f s0 = zero, s1 = zero;
    v8bf lo0 = *(const v8bf*)&kRow[ln * D_DIM + 8 * h];
    v8bf hi0 = *(const v8bf*)&kRow[ln * D_DIM + 8 * h + 16];
    v8bf lo1 = *(const v8bf*)&kRow[(16 + ln) * D_DIM + 8 * h];
    v8bf hi1 = *(const v8bf*)&kRow[(16 + ln) * D_DIM + 8 * h + 16];
#pragma unroll
    for (int c = 0; c < 8; ++c) {
      v8bf nlo0, nhi0, nlo1, nhi1;
      if (c < 7) {   // issue next iteration's loads before consuming current
        const int off = (c + 1) * 32 + 8 * h;
        nlo0 = *(const v8bf*)&kRow[ln * D_DIM + off];
        nhi0 = *(const v8bf*)&kRow[ln * D_DIM + off + 16];
        nlo1 = *(const v8bf*)&kRow[(16 + ln) * D_DIM + off];
        nhi1 = *(const v8bf*)&kRow[(16 + ln) * D_DIM + off + 16];
      }
      v16bf a0, a1;
#pragma unroll
      for (int e = 0; e < 8; ++e) {
        a0[e] = lo0[e]; a0[e + 8] = hi0[e];
        a1[e] = lo1[e]; a1[e + 8] = hi1[e];
      }
      s0 = wmma_bf16(a0, qfrag[c], s0);
      s1 = wmma_bf16(a1, qfrag[c], s1);
      lo0 = nlo0; hi0 = nhi0; lo1 = nlo1; hi1 = nhi1;
    }
    // Sᵀ lane(h,n): query col n, key row m = v + 8h (tile0: keys kb.., tile1: kb+16..)

    // -------- online softmax over m (per-lane + one xor-16 combine) ---------
    float lmax = -1e30f;
#pragma unroll
    for (int v = 0; v < 8; ++v) lmax = fmaxf(lmax, fmaxf(s0[v], s1[v]));
    lmax = fmaxf(lmax, __shfl_xor(lmax, 16, 32));
    const float nm     = fmaxf(run_max, lmax);
    const float nml    = nm * LOG2E;
    const float factor = exp2f((run_max - nm) * LOG2E);

    float p0[8], p1[8];
    float lsum = 0.0f;
#pragma unroll
    for (int v = 0; v < 8; ++v) {
      p0[v] = exp2f(fmaf(s0[v], LOG2E, -nml));
      p1[v] = exp2f(fmaf(s1[v], LOG2E, -nml));
      lsum += p0[v] + p1[v];
    }
    run_sum = fmaf(run_sum, factor, lsum);

#if __has_builtin(__builtin_amdgcn_ballot_w32)
    const bool grew = __builtin_amdgcn_ballot_w32(lmax > run_max) != 0u;
#else
    const bool grew = __any(lmax > run_max) != 0;
#endif
    run_max = nm;
    if (grew) {   // wave-uniform: skipped once the running max stabilizes
#pragma unroll
      for (int t = 0; t < 16; ++t)
#pragma unroll
        for (int v = 0; v < 8; ++v) acc[t][v] *= factor;
    }

    // -------- pack Pᵀ into B-operand layout (one shuffle per VGPR) ----------
    // B needs lane(h,n): chunk keys K=16h+e, e=0..15. Own data covers m=8h+v of
    // each tile; the missing half lives in the partner lane (xor 16).
    float mlow[8], mhigh[8];
#pragma unroll
    for (int v = 0; v < 8; ++v) {
      const float send = h ? p0[v] : p1[v];
      const float r = __shfl_xor(send, 16, 32);
      mlow[v]  = h ? r : p0[v];      // tile h, m = v
      mhigh[v] = h ? p1[v] : r;      // tile h, m = 8 + v
    }
    v16bf bP;
#pragma unroll
    for (int i = 0; i < 4; ++i) {
      const v2bf plo = f2bf2(mlow[2 * i], mlow[2 * i + 1]);
      const v2bf phi = f2bf2(mhigh[2 * i], mhigh[2 * i + 1]);
      bP[2 * i]     = plo[0]; bP[2 * i + 1]     = plo[1];
      bP[8 + 2 * i] = phi[0]; bP[8 + 2 * i + 1] = phi[1];
    }

    // -------- Outᵀ += Kᵀ(16d x 32keys) x Pᵀ(32keys x 16q), pipelined --------
    v8bf tlo = *(const v8bf*)&kT[ln * CHUNK + 16 * h];
    v8bf thi = *(const v8bf*)&kT[ln * CHUNK + 16 * h + 8];
#pragma unroll
    for (int t = 0; t < 16; ++t) {
      v8bf ntlo, nthi;
      if (t < 15) {
        const int drow = (t + 1) * 16 + ln;
        ntlo = *(const v8bf*)&kT[drow * CHUNK + 16 * h];
        nthi = *(const v8bf*)&kT[drow * CHUNK + 16 * h + 8];
      }
      v16bf aK;
#pragma unroll
      for (int e = 0; e < 8; ++e) { aK[e] = tlo[e]; aK[e + 8] = thi[e]; }
      acc[t] = wmma_bf16(aK, bP, acc[t]);
      tlo = ntlo; thi = nthi;
    }

    // -------- convert + store next chunk into the other LDS buffer ----------
    if (more) {
      __bf16* rowDst = kRowB[p ^ 1];
      __bf16* tDst   = kTB[p ^ 1];
#pragma unroll
      for (int i = 0; i < 16; ++i) {
        const float4 v = g[i];
        const int d = d0s + 4 * i;
        const v2bf c01 = f2bf2(v.x, v.y);
        const v2bf c23 = f2bf2(v.z, v.w);
        v4bf row; row[0] = c01[0]; row[1] = c01[1]; row[2] = c23[0]; row[3] = c23[1];
        *(v4bf*)&rowDst[kk * D_DIM + d] = row;
        tDst[(d + 0) * CHUNK + kk] = c01[0];
        tDst[(d + 1) * CHUNK + kk] = c01[1];
        tDst[(d + 2) * CHUNK + kk] = c23[0];
        tDst[(d + 3) * CHUNK + kk] = c23[1];
      }
    }
    p ^= 1;
  }

  // ---------------- epilogue: normalize and store NCHW ----------------------
  const float tot = run_sum + __shfl_xor(run_sum, 16, 32);
  const float inv = 1.0f / tot;
  const int orow = wave_row0 + ln;        // this lane's query (D-tile column)
  const int ob   = orow >> 12;
  const int ohw  = orow & (HW - 1);
  float* ocol = out + (size_t)ob * (D_DIM * HW) + ohw;
#pragma unroll
  for (int t = 0; t < 16; ++t) {
#pragma unroll
    for (int v = 0; v < 8; ++v) {
      const int d = t * 16 + 8 * h + v;   // Outᵀ row M = v + 8h within tile t
      ocol[(size_t)d * HW] = acc[t][v] * inv;
    }
  }
}

extern "C" void kernel_launch(void* const* d_in, const int* in_sizes, int n_in,
                              void* d_out, int out_size, void* d_ws, size_t ws_size,
                              hipStream_t stream) {
  (void)n_in; (void)out_size; (void)d_ws; (void)ws_size;
  const float* keys  = (const float*)d_in[0];   // [2048, 256] f32
  const float* query = (const float*)d_in[1];   // [8, 256, 64, 64] f32
  float* out = (float*)d_out;                   // [8, 256, 64, 64] f32

  const int n_rows = in_sizes[1] / D_DIM;       // B*H*W = 32768
  const int n_wg   = n_rows / ROWS_PER_WG;      // 512
  dim3 grid(n_wg), block(WAVES_PER_WG * 32);
  hipLaunchKernelGGL(mem_attn_read_kernel, grid, block, 0, stream,
                     keys, query, out);
}
